// DeformableTransformerDecoders_10136122819001
// MI455X (gfx1250) — matrix operand order repaired and checked by reading, not compiled, not run on hardware
//
#include <hip/hip_runtime.h>
#include <hip/hip_bf16.h>
#include <math.h>

typedef __attribute__((ext_vector_type(16))) _Float16 v16h;
typedef __attribute__((ext_vector_type(8)))  float    v8f;

#define WMMA16(a, b, c) __builtin_amdgcn_wmma_f32_16x16x32_f16(false, (a), false, (b), (short)0, (c), false, false)

// ---------------------------------------------------------------------------
// Fragment loaders (CDNA5 16-bit WMMA VGPR layouts, wave32)
// A 16x32 (MxK): lane L -> m = L%16, g = L/16; elems 0..7 -> k = 8g+e,
//                elems 8..15 -> k = 16+8g+(e-8)  (two contiguous 8-half runs)
// B 32x16 (KxN): lane L -> n = L%16, g = L/16; same k order as A (two 8-half runs)
// ---------------------------------------------------------------------------
__device__ __forceinline__ v16h ld_a_frag(const _Float16* base, int row0, int stride,
                                          int koff = 0) {
  int lane = (int)(threadIdx.x & 31);
  int g = lane >> 4, mm = lane & 15;
  const _Float16* r = base + (size_t)(row0 + mm) * stride + koff;
  union { v16h v; uint4 u[2]; } t;
  t.u[0] = *reinterpret_cast<const uint4*>(r + 8 * g);
  t.u[1] = *reinterpret_cast<const uint4*>(r + 16 + 8 * g);
  return t.v;
}

__device__ __forceinline__ v16h ld_b_frag(const _Float16* base, int n0, int stride,
                                          int koff = 0) {
  int lane = (int)(threadIdx.x & 31);
  int g = lane >> 4, nn = lane & 15;
  const _Float16* r = base + (size_t)(n0 + nn) * stride + koff;
  union { v16h v; uint4 u[2]; } t;
  t.u[0] = *reinterpret_cast<const uint4*>(r + 8 * g);
  t.u[1] = *reinterpret_cast<const uint4*>(r + 16 + 8 * g);
  return t.v;
}

// ---------------------------------------------------------------------------
// GEMM: C[M,N] = A[M,K] @ W[N,K]^T + bias, optional ReLU.
// Block 256 thr = 8 waves (4x2), tile 128x64, wave tile 32x32, K-step 64
// (two WMMA sub-steps -> 8 WMMAs per wave per barrier window).
// Requires M%128==0, N%64==0, K%64==0 (always true here).
// ---------------------------------------------------------------------------
__global__ __launch_bounds__(256) void gemm_wmma_kernel(
    const float* __restrict__ A, const float* __restrict__ W,
    const float* __restrict__ bias, float* __restrict__ C,
    int M, int N, int K, int relu) {
  __shared__ _Float16 As[128 * 72];   // 128 rows x 64 halves, stride 72 (144B)
  __shared__ _Float16 Bs[64 * 72];
  int tid = (int)threadIdx.x;
  int wave = tid >> 5;
  int wm = wave >> 1, wn = wave & 1;
  int bm = (int)blockIdx.x * 128, bn = (int)blockIdx.y * 64;

  v8f acc[2][2] = {};

  for (int k0 = 0; k0 < K; k0 += 64) {
    __syncthreads();
    { // A tile 128x64 -> fp16 LDS (32 floats per thread)
      int r = tid >> 1, cc = (tid & 1) * 32;
      const float* sp = A + (size_t)(bm + r) * K + k0 + cc;
      _Float16* dp = As + r * 72 + cc;
#pragma unroll
      for (int i = 0; i < 32; i += 4) {
        float4 f = *reinterpret_cast<const float4*>(sp + i);
        dp[i + 0] = (_Float16)f.x; dp[i + 1] = (_Float16)f.y;
        dp[i + 2] = (_Float16)f.z; dp[i + 3] = (_Float16)f.w;
      }
      if (k0 + 64 < K) __builtin_prefetch(sp + 64, 0, 0);   // next K-tile
    }
    { // B tile (W rows) 64x64 -> fp16 LDS (16 floats per thread)
      int r = tid >> 2, cc = (tid & 3) * 16;
      const float* sp = W + (size_t)(bn + r) * K + k0 + cc;
      _Float16* dp = Bs + r * 72 + cc;
#pragma unroll
      for (int i = 0; i < 16; i += 4) {
        float4 f = *reinterpret_cast<const float4*>(sp + i);
        dp[i + 0] = (_Float16)f.x; dp[i + 1] = (_Float16)f.y;
        dp[i + 2] = (_Float16)f.z; dp[i + 3] = (_Float16)f.w;
      }
      if (k0 + 64 < K) __builtin_prefetch(sp + 64, 0, 0);
    }
    __syncthreads();
#pragma unroll
    for (int ks = 0; ks < 64; ks += 32) {
      v16h a0 = ld_a_frag(As, wm * 32, 72, ks);
      v16h a1 = ld_a_frag(As, wm * 32 + 16, 72, ks);
      v16h b0 = ld_b_frag(Bs, wn * 32, 72, ks);
      v16h b1 = ld_b_frag(Bs, wn * 32 + 16, 72, ks);
      acc[0][0] = WMMA16(a0, b0, acc[0][0]);
      acc[0][1] = WMMA16(a0, b1, acc[0][1]);
      acc[1][0] = WMMA16(a1, b0, acc[1][0]);
      acc[1][1] = WMMA16(a1, b1, acc[1][1]);
    }
  }

  // C layout: vgpr r -> m = r + 8*(lane/16); n = lane%16
  int lane = tid & 31, g = lane >> 4, nn = lane & 15;
#pragma unroll
  for (int i = 0; i < 2; i++) {
#pragma unroll
    for (int j = 0; j < 2; j++) {
      int ncol = bn + wn * 32 + j * 16 + nn;
      float bv = bias[ncol];
#pragma unroll
      for (int r = 0; r < 8; r++) {
        int mrow = bm + wm * 32 + i * 16 + 8 * g + r;
        float v = acc[i][j][r] + bv;
        if (relu) v = fmaxf(v, 0.0f);
        C[(size_t)mrow * N + ncol] = v;
      }
    }
  }
}

// ---------------------------------------------------------------------------
// Fused flash self-attention. qkbuf: [B*2048, 512] (q cols 0..255, k cols
// 256..511 per head h at h*32). vbuf: [B*2048, 256]. out: [B*2048, 256].
// grid (B*NH, 2048/64), block 128 = 4 waves x 16 queries, Dh=32.
// ---------------------------------------------------------------------------
__global__ __launch_bounds__(128) void attn_flash_kernel(
    const float* __restrict__ qkbuf, const float* __restrict__ vbuf,
    float* __restrict__ out) {
  __shared__ _Float16 Ks[32 * 40];       // [key][ch]
  __shared__ _Float16 Vt[32 * 40];       // [ch][key] (transposed)
  __shared__ _Float16 Ps[4 * 16 * 40];   // per-wave P staging

  int b = (int)blockIdx.x >> 3;          // / NH(8)
  int h = (int)blockIdx.x & 7;
  int tid = (int)threadIdx.x;
  int wave = tid >> 5, lane = tid & 31, g = lane >> 4, nn = lane & 15;
  int qbase = (int)blockIdx.y * 64 + wave * 16;
  size_t rowb = (size_t)b * 2048;

  // Q fragment (A layout), pre-scaled by 1/sqrt(Dh)
  union { v16h v; _Float16 e[16]; } qf;
  {
    const float sc = 0.17677669529663687f;
    const float* qp = qkbuf + (rowb + qbase + nn) * 512 + h * 32;
#pragma unroll
    for (int e = 0; e < 8; e++) qf.e[e]     = (_Float16)(qp[8 * g + e] * sc);
#pragma unroll
    for (int e = 0; e < 8; e++) qf.e[8 + e] = (_Float16)(qp[16 + 8 * g + e] * sc);
  }

  v8f o0 = {}, o1 = {};
  float lsum[8], mx[8];
#pragma unroll
  for (int i = 0; i < 8; i++) { lsum[i] = 0.0f; mx[i] = -1e30f; }

  v8f zero = {};
  _Float16* pw = Ps + wave * 16 * 40;

  for (int kt = 0; kt < 2048; kt += 32) {
    __syncthreads();
    { // cooperative K / V^T tile loads (fp32 -> fp16)
      int key = tid >> 2, cc = (tid & 3) * 8;
      const float* ks = qkbuf + (rowb + kt + key) * 512 + 256 + h * 32 + cc;
      _Float16* kd = Ks + key * 40 + cc;
#pragma unroll
      for (int i = 0; i < 8; i++) kd[i] = (_Float16)ks[i];
      const float* vs = vbuf + (rowb + kt + key) * 256 + h * 32 + cc;
#pragma unroll
      for (int i = 0; i < 8; i++) Vt[(cc + i) * 40 + key] = (_Float16)vs[i];
      if (kt + 32 < 2048) {
        __builtin_prefetch(ks + 32 * 512, 0, 0);
        __builtin_prefetch(vs + 32 * 256, 0, 0);
      }
    }
    __syncthreads();

    v16h kb0 = ld_b_frag(Ks, 0, 40);
    v16h kb1 = ld_b_frag(Ks, 16, 40);
    v8f s0 = WMMA16(qf.v, kb0, zero);
    v8f s1 = WMMA16(qf.v, kb1, zero);

    // online softmax over the 32 new keys
#pragma unroll
    for (int i = 0; i < 8; i++) {
      float v0 = s0[i], v1 = s1[i];
      float rmax = fmaxf(v0, v1);
      rmax = fmaxf(rmax, __shfl_xor(rmax, 1, 32));
      rmax = fmaxf(rmax, __shfl_xor(rmax, 2, 32));
      rmax = fmaxf(rmax, __shfl_xor(rmax, 4, 32));
      rmax = fmaxf(rmax, __shfl_xor(rmax, 8, 32));
      float mnew = fmaxf(mx[i], rmax);
      float scold = __expf(mx[i] - mnew);
      float p0 = __expf(v0 - mnew);
      float p1 = __expf(v1 - mnew);
      float rs = p0 + p1;
      rs += __shfl_xor(rs, 1, 32);
      rs += __shfl_xor(rs, 2, 32);
      rs += __shfl_xor(rs, 4, 32);
      rs += __shfl_xor(rs, 8, 32);
      lsum[i] = lsum[i] * scold + rs;
      mx[i] = mnew;
      o0[i] *= scold;
      o1[i] *= scold;
      int m = i + 8 * g;
      pw[m * 40 + nn]      = (_Float16)p0;
      pw[m * 40 + 16 + nn] = (_Float16)p1;
    }
    __syncthreads();

    v16h pf  = ld_a_frag(pw, 0, 40);
    v16h vb0 = ld_b_frag(Vt, 0, 40);
    v16h vb1 = ld_b_frag(Vt, 16, 40);
    o0 = WMMA16(pf, vb0, o0);
    o1 = WMMA16(pf, vb1, o1);
  }

#pragma unroll
  for (int i = 0; i < 8; i++) {
    int m = i + 8 * g;
    float inv = 1.0f / lsum[i];
    float* op = out + (rowb + qbase + m) * 256 + h * 32;
    op[nn]      = o0[i] * inv;
    op[16 + nn] = o1[i] * inv;
  }
}

// ---------------------------------------------------------------------------
// Elementwise / reduction helpers
// ---------------------------------------------------------------------------
__global__ __launch_bounds__(256) void add_kernel(
    const float* __restrict__ a, const float* __restrict__ b,
    float* __restrict__ o, int n) {
  int i = (int)blockIdx.x * 256 + (int)threadIdx.x;
  if (i < n) o[i] = a[i] + b[i];
}

// out = LN(x + d) * g + b   (one 256-wide row per block; safe in-place)
__global__ __launch_bounds__(256) void resln_kernel(
    const float* __restrict__ x, const float* __restrict__ dlt,
    const float* __restrict__ gam, const float* __restrict__ bet,
    float* __restrict__ out) {
  __shared__ float red[256];
  int row = (int)blockIdx.x, t = (int)threadIdx.x;
  size_t base = (size_t)row * 256;
  float v = x[base + t] + dlt[base + t];
  red[t] = v;
  __syncthreads();
  for (int s = 128; s > 0; s >>= 1) { if (t < s) red[t] += red[t + s]; __syncthreads(); }
  float mean = red[0] * (1.0f / 256.0f);
  __syncthreads();
  float d = v - mean;
  red[t] = d * d;
  __syncthreads();
  for (int s = 128; s > 0; s >>= 1) { if (t < s) red[t] += red[t + s]; __syncthreads(); }
  float var = red[0] * (1.0f / 256.0f);
  out[base + t] = d * rsqrtf(var + 1e-5f) * gam[t] + bet[t];
}

// softmax over contiguous groups of 8 (aw over NL*NP per head)
__global__ __launch_bounds__(256) void awsm_kernel(float* __restrict__ aw, int ngroups) {
  int gi = (int)blockIdx.x * 256 + (int)threadIdx.x;
  if (gi >= ngroups) return;
  float* p = aw + (size_t)gi * 8;
  float m = -1e30f;
#pragma unroll
  for (int i = 0; i < 8; i++) m = fmaxf(m, p[i]);
  float e[8], s = 0.0f;
#pragma unroll
  for (int i = 0; i < 8; i++) { e[i] = __expf(p[i] - m); s += e[i]; }
  float inv = 1.0f / s;
#pragma unroll
  for (int i = 0; i < 8; i++) p[i] = e[i] * inv;
}

// Deformable bilinear sampling; one 32-lane group per (b,q,h), lane = channel d
__global__ __launch_bounds__(256) void msda_sample_kernel(
    const float* __restrict__ val, const float* __restrict__ off,
    const float* __restrict__ aw, float* __restrict__ out) {
  int unit = (int)blockIdx.x * 8 + ((int)threadIdx.x >> 5);
  int d = (int)threadIdx.x & 31;
  int h = unit & 7;             // % NH
  int mq = unit >> 3;           // b*2048 + q
  int b = mq >> 11;
  int q = mq & 2047;
  int inframe = q & 1023;
  int row = inframe >> 5, col = inframe & 31;
  float rx = (col + 0.5f) * (1.0f / 32.0f);
  float ry = (row + 0.5f) * (1.0f / 32.0f);

  const float* offp = off + (size_t)mq * 128 + h * 16;
  const float* awp  = aw  + (size_t)mq * 64  + h * 8;
  float acc = 0.0f;

#pragma unroll
  for (int lvl = 0; lvl < 2; lvl++) {
    const float* vb = val + ((size_t)b * 2048 + lvl * 1024) * 256 + h * 32 + d;
#pragma unroll
    for (int p = 0; p < 4; p++) {
      float ox = offp[lvl * 8 + p * 2 + 0];
      float oy = offp[lvl * 8 + p * 2 + 1];
      float w  = awp[lvl * 4 + p];
      float x = (rx + ox * (1.0f / 32.0f)) * 32.0f - 0.5f;
      float y = (ry + oy * (1.0f / 32.0f)) * 32.0f - 0.5f;
      float x0f = floorf(x), y0f = floorf(y);
      int x0 = (int)x0f, y0 = (int)y0f;
      float fx = x - x0f, fy = y - y0f;
      float w00 = (1.0f - fx) * (1.0f - fy);
      float w01 = fx * (1.0f - fy);
      float w10 = (1.0f - fx) * fy;
      float w11 = fx * fy;
      if (x0 >= 0 && x0 < 32 && y0 >= 0 && y0 < 32)
        acc += w * w00 * vb[(size_t)(y0 * 32 + x0) * 256];
      if (x0 + 1 >= 0 && x0 + 1 < 32 && y0 >= 0 && y0 < 32)
        acc += w * w01 * vb[(size_t)(y0 * 32 + x0 + 1) * 256];
      if (x0 >= 0 && x0 < 32 && y0 + 1 >= 0 && y0 + 1 < 32)
        acc += w * w10 * vb[(size_t)((y0 + 1) * 32 + x0) * 256];
      if (x0 + 1 >= 0 && x0 + 1 < 32 && y0 + 1 >= 0 && y0 + 1 < 32)
        acc += w * w11 * vb[(size_t)((y0 + 1) * 32 + x0 + 1) * 256];
    }
  }
  out[(size_t)mq * 256 + h * 32 + d] = acc;
}

// reference points output: (1, 2048, 2, 2)
__global__ __launch_bounds__(256) void ref_kernel(float* __restrict__ o) {
  int i = (int)blockIdx.x * 256 + (int)threadIdx.x;
  if (i >= 2048) return;
  int f = i & 1023;
  int row = f >> 5, col = f & 31;
  float rx = (col + 0.5f) * (1.0f / 32.0f);
  float ry = (row + 0.5f) * (1.0f / 32.0f);
#pragma unroll
  for (int lvl = 0; lvl < 2; lvl++) {
    o[(size_t)(i * 2 + lvl) * 2 + 0] = rx;
    o[(size_t)(i * 2 + lvl) * 2 + 1] = ry;
  }
}

// ---------------------------------------------------------------------------
// Host orchestration
// ---------------------------------------------------------------------------
static inline void launch_gemm(const float* A, const float* W, const float* bias,
                               float* C, int M, int N, int K, int relu,
                               hipStream_t stream) {
  dim3 grid(M / 128, N / 64);
  gemm_wmma_kernel<<<grid, 256, 0, stream>>>(A, W, bias, C, M, N, K, relu);
}

extern "C" void kernel_launch(void* const* d_in, const int* in_sizes, int n_in,
                              void* d_out, int out_size, void* d_ws, size_t ws_size,
                              hipStream_t stream) {
  (void)in_sizes; (void)n_in; (void)out_size; (void)ws_size;
  const float* src     = (const float*)d_in[0];
  const float* tgt     = (const float*)d_in[1];
  const float* src_pos = (const float*)d_in[2];
  const float* tgt_pos = (const float*)d_in[3];
  const float* sa_w    = (const float*)d_in[4];
  const float* sa_b    = (const float*)d_in[5];
  const float* sao_w   = (const float*)d_in[6];
  const float* sao_b   = (const float*)d_in[7];
  const float* off_w   = (const float*)d_in[8];
  const float* off_b   = (const float*)d_in[9];
  const float* aw_w    = (const float*)d_in[10];
  const float* aw_b    = (const float*)d_in[11];
  const float* val_w   = (const float*)d_in[12];
  const float* val_b   = (const float*)d_in[13];
  const float* cao_w   = (const float*)d_in[14];
  const float* cao_b   = (const float*)d_in[15];
  const float* ln1_g   = (const float*)d_in[16];
  const float* ln1_b   = (const float*)d_in[17];
  const float* ln2_g   = (const float*)d_in[18];
  const float* ln2_b   = (const float*)d_in[19];
  const float* ln3_g   = (const float*)d_in[20];
  const float* ln3_b   = (const float*)d_in[21];
  const float* ff1_w   = (const float*)d_in[22];
  const float* ff1_b   = (const float*)d_in[23];
  const float* ff2_w   = (const float*)d_in[24];
  const float* ff2_b   = (const float*)d_in[25];

  const int M = 4096;                       // B * T*Hs*Ws tokens
  const size_t M1 = (size_t)M * 256;        // 1,048,576 floats

  float* ws  = (float*)d_ws;
  float* x   = ws;                          // (M,256)
  float* sv  = ws + 1 * M1;                 // (M,256)
  float* qb  = ws + 2 * M1;                 // (M,256)  q / q2 / samp
  float* qk  = ws + 3 * M1;                 // (M,512); hbuf reuses qk..ao span
  float* vb  = ws + 5 * M1;                 // (M,256)
  float* ao  = ws + 6 * M1;                 // (M,256)
  float* t1  = ws + 7 * M1;                 // (M,256)
  float* off = ws + 8 * M1;                 // (M,128)
  float* awb = off + (size_t)M * 128;       // (M,64)
  float* val = awb + (size_t)M * 64;        // (M,256)
  float* hb  = qk;                          // (M,1024) overlaps qk+vb+ao (dead)

  hipMemcpyAsync(x, tgt, M1 * sizeof(float), hipMemcpyDeviceToDevice, stream);
  add_kernel<<<(int)(M1 / 256), 256, 0, stream>>>(src, src_pos, sv, (int)M1);

  for (int l = 0; l < 6; l++) {
    // ---- self attention ----
    add_kernel<<<(int)(M1 / 256), 256, 0, stream>>>(x, tgt_pos, qb, (int)M1);
    launch_gemm(qb, sa_w + (size_t)l * 196608, sa_b + (size_t)l * 768,
                qk, M, 512, 256, 0, stream);                           // Q,K proj
    launch_gemm(x, sa_w + (size_t)l * 196608 + 131072, sa_b + (size_t)l * 768 + 512,
                vb, M, 256, 256, 0, stream);                           // V proj
    attn_flash_kernel<<<dim3(16, 32), 128, 0, stream>>>(qk, vb, ao);
    launch_gemm(ao, sao_w + (size_t)l * 65536, sao_b + (size_t)l * 256,
                t1, M, 256, 256, 0, stream);                           // out proj
    resln_kernel<<<M, 256, 0, stream>>>(x, t1, ln2_g + l * 256, ln2_b + l * 256, x);

    // ---- MSDA cross attention ----
    add_kernel<<<(int)(M1 / 256), 256, 0, stream>>>(x, tgt_pos, qb, (int)M1);
    launch_gemm(sv, val_w + (size_t)l * 65536, val_b + (size_t)l * 256,
                val, M, 256, 256, 0, stream);                          // value proj
    launch_gemm(qb, off_w + (size_t)l * 32768, off_b + (size_t)l * 128,
                off, M, 128, 256, 0, stream);                          // offsets
    launch_gemm(qb, aw_w + (size_t)l * 16384, aw_b + (size_t)l * 64,
                awb, M, 64, 256, 0, stream);                           // attn weights
    awsm_kernel<<<128, 256, 0, stream>>>(awb, M * 8);
    msda_sample_kernel<<<M, 256, 0, stream>>>(val, off, awb, qb);
    launch_gemm(qb, cao_w + (size_t)l * 65536, cao_b + (size_t)l * 256,
                t1, M, 256, 256, 0, stream);                           // out proj
    resln_kernel<<<M, 256, 0, stream>>>(x, t1, ln1_g + l * 256, ln1_b + l * 256, x);

    // ---- FFN ----
    launch_gemm(x, ff1_w + (size_t)l * 262144, ff1_b + (size_t)l * 1024,
                hb, M, 1024, 256, 1, stream);                          // 256->1024 +ReLU
    launch_gemm(hb, ff2_w + (size_t)l * 262144, ff2_b + (size_t)l * 256,
                t1, M, 256, 1024, 0, stream);                          // 1024->256
    resln_kernel<<<M, 256, 0, stream>>>(x, t1, ln3_g + l * 256, ln3_b + l * 256, x);
  }

  float* outp = (float*)d_out;
  hipMemcpyAsync(outp, x, M1 * sizeof(float), hipMemcpyDeviceToDevice, stream);
  ref_kernel<<<8, 256, 0, stream>>>(outp + M1);
}
